// DeformConvWithOff_11982958756208
// MI455X (gfx1250) — matrix operand order repaired and checked by reading, not compiled, not run on hardware
//
#include <hip/hip_runtime.h>
#include <hip/hip_bf16.h>

// ---------------- CDNA5 vector types ----------------
typedef _Float16 v16h __attribute__((ext_vector_type(16)));
typedef _Float16 v8h  __attribute__((ext_vector_type(8)));
typedef float    v8f  __attribute__((ext_vector_type(8)));

// ---------------- problem constants ----------------
#define BB    8
#define CC    256
#define HH    64
#define WW    64
#define TAPS  9
#define KD    2304      // CC*TAPS : GEMM K
#define MT    128       // M tile per workgroup
#define KS    32        // K step (wmma f16 depth)
#define LDA   40        // LDS row stride (halfs): 80B = 20 banks -> conflict-free b128 frag reads
#define MTOT  32768     // BB*HH*WW

// LDS partition (bytes); disasm confirmed dynamic-LDS base == 0, so these
// byte offsets are also the absolute LDS addresses used by async loads.
#define SAMP_W_BYTES  (MT*TAPS*16)              // float4 bilinear weights  = 18432
#define SAMP_I_BYTES  (MT*TAPS*8)               // uint2 packed gather byte-offsets = 9216
#define SAMP_BYTES    (SAMP_W_BYTES + SAMP_I_BYTES)   // 27648
#define A_BYTES       (MT*LDA*2)                // 10240
#define B_LDS_BASE    (SAMP_BYTES + A_BYTES)    // 37888

static __device__ __forceinline__ v16h cat8(v8h a, v8h b) {
  return __builtin_shufflevector(a, b, 0,1,2,3,4,5,6,7,8,9,10,11,12,13,14,15);
}
static __device__ __forceinline__ v8f vzero8() {
  v8f z = {0.f,0.f,0.f,0.f,0.f,0.f,0.f,0.f};
  return z;
}

// -------- weight prep: f32 -> f16 (w_off zero-padded to 32 rows) --------
__global__ void cvt_pad_woff(const float* __restrict__ w, _Float16* __restrict__ o) {
  int i = blockIdx.x * 256 + threadIdx.x;
  if (i < 32 * KD) {
    int r = i / KD;
    o[i] = (r < 18) ? (_Float16)w[i] : (_Float16)0.f;
  }
}
__global__ void cvt_wdcn(const float* __restrict__ w, _Float16* __restrict__ o, int n) {
  int i = blockIdx.x * 256 + threadIdx.x;
  if (i < n) o[i] = (_Float16)w[i];
}

// ---------------------------------------------------------------------------
// Fused implicit-im2col (+ bilinear deform sampling) f16 WMMA GEMM,
// register-level software pipelined:
//   stage tile t -> load frags(t) -> barrier ->
//   [async B(t+1) | WMMA(t) on matrix pipe | bilinear A-fill(t+1) on VALU]
// ---------------------------------------------------------------------------
template<int NT, bool DEFORM>
__global__ __launch_bounds__(256)
void dcn_wmma_gemm(const float* __restrict__ x,
                   const float* __restrict__ off,        // (B,18,64,64) f32, DEFORM only
                   const _Float16* __restrict__ Wt,      // [>=NT rows][KD] f16
                   const float* __restrict__ bias,       // [Nvalid], !DEFORM only
                   float* __restrict__ out, int Nvalid)
{
  extern __shared__ char smem[];
  float4*   sw4 = (float4*)smem;                           // [MT*TAPS] bilinear weights
  uint2*    sij = (uint2*)(smem + SAMP_W_BYTES);           // [MT*TAPS] packed byte offsets
  _Float16* As  = (_Float16*)(smem + SAMP_BYTES);          // [MT][LDA]
  _Float16* Bs  = (_Float16*)(smem + B_LDS_BASE);          // [NT][LDA]

  const int tid = threadIdx.x;
  const int m0  = blockIdx.x * MT;
  const int n0  = blockIdx.y * NT;
  const int bb  = m0 >> 12;          // scalar batch index for whole tile
  const int hw0 = m0 & 4095;         // scalar hw base for whole tile
  const char* __restrict__ xB = (const char*)(x + ((size_t)bb << 20)); // x[bb,0,0,0]

  // ---- Phase 0: per-(position,tap) bilinear metadata (shared by all 256 c) ----
  for (int i = tid; i < MT * TAPS; i += 256) {
    int r   = i / TAPS;
    int tap = i - TAPS * r;
    int hw  = hw0 + r;
    int h   = hw >> 6, w = hw & 63;
    float oy = 0.f, ox = 0.f;
    if constexpr (DEFORM) {
      int ob = (bb * 18 + tap * 2) * 4096 + hw;
      oy = off[ob];
      ox = off[ob + 4096];
    }
    float sy = (float)(h - 1 + tap / 3) + oy;
    float sx = (float)(w - 1 + tap % 3) + ox;
    float y0f = floorf(sy), x0f = floorf(sx);
    float fy = sy - y0f, fx = sx - x0f;
    int y0 = (int)y0f, x0 = (int)x0f;
    int y1 = y0 + 1,   x1 = x0 + 1;
    bool vy0 = (y0 >= 0) & (y0 < HH), vy1 = (y1 >= 0) & (y1 < HH);
    bool vx0 = (x0 >= 0) & (x0 < WW), vx1 = (x1 >= 0) & (x1 < WW);
    float w00 = (1.f - fy) * (1.f - fx) * (float)(vy0 & vx0);
    float w01 = (1.f - fy) * fx         * (float)(vy0 & vx1);
    float w10 = fy * (1.f - fx)         * (float)(vy1 & vx0);
    float w11 = fy * fx                 * (float)(vy1 & vx1);
    unsigned yc0 = (unsigned)min(max(y0, 0), HH - 1);
    unsigned yc1 = (unsigned)min(max(y1, 0), HH - 1);
    unsigned xc0 = (unsigned)min(max(x0, 0), WW - 1);
    unsigned xc1 = (unsigned)min(max(x1, 0), WW - 1);
    // pre-scaled BYTE offsets (max 4095*4 = 16380, fits u16)
    unsigned i00 = (yc0 * WW + xc0) * 4u, i01 = (yc0 * WW + xc1) * 4u;
    unsigned i10 = (yc1 * WW + xc0) * 4u, i11 = (yc1 * WW + xc1) * 4u;
    sw4[i] = make_float4(w00, w01, w10, w11);
    sij[i] = make_uint2(i00 | (i01 << 16), i10 | (i11 << 16));
  }

  // ---- wave tiling ----
  constexpr int WVM = (NT == 256) ? 2 : 8;  // waves along M
  constexpr int WVN = 8 / WVM;              // waves along N
  constexpr int MF  = MT / (16 * WVM);      // 16x16 m-frags per wave
  constexpr int NF  = NT / (16 * WVN);      // 16x16 n-frags per wave
  const int wv   = tid >> 5;
  const int lane = tid & 31;
  const int lr   = lane & 15;
  const int hi   = lane >> 4;
  const int wm   = (wv % WVM) * (MF * 16);
  const int wn   = (wv / WVM) * (NF * 16);

  v8f acc[MF][NF];
#pragma unroll
  for (int mt = 0; mt < MF; ++mt)
#pragma unroll
    for (int nt = 0; nt < NF; ++nt) acc[mt][nt] = vzero8();

  const unsigned long long wt64 = (unsigned long long)Wt;

  v16h aF[MF], bF[NF];

  // B tile: async DMA memory -> LDS, 16B per lane (ASYNCcnt-tracked)
  auto asyncB = [&](int k0) {
    for (int j = tid; j < NT * KS / 8; j += 256) {
      int row = j >> 2;
      int cp  = (j & 3) * 8;
      unsigned ldsoff = (unsigned)(B_LDS_BASE) + (unsigned)(row * LDA + cp) * 2u;
      unsigned goff   = (unsigned)(((n0 + row) * KD + k0 + cp) * 2);
      asm volatile("global_load_async_to_lds_b128 %0, %1, %2"
                   :: "v"(ldsoff), "v"(goff), "s"(wt64) : "memory");
    }
  };

  // A tile: fused bilinear im2col; rows r and r+64 share (c,tap) work.
  // All 8 corner gathers issued before any FMA -> deep MLP, no per-corner waits.
  auto fillA = [&](int k0) {
#pragma unroll
    for (int i = tid; i < MT * KS / 2; i += 256) {
      int r   = i >> 5;          // 0..63
      int col = i & 31;
      int kk  = k0 + col;
      int c   = kk / 9;
      int tap = kk - 9 * c;
      const char* __restrict__ xb = xB + ((size_t)c << 14);   // + c*4096*4 bytes
      float4 wa  = sw4[r * TAPS + tap];
      uint2  ja  = sij[r * TAPS + tap];
      float4 wb2 = sw4[(r + 64) * TAPS + tap];
      uint2  jb  = sij[(r + 64) * TAPS + tap];
      float a0 = *(const float*)(xb + (ja.x & 0xffffu));
      float a1 = *(const float*)(xb + (ja.x >> 16));
      float a2 = *(const float*)(xb + (ja.y & 0xffffu));
      float a3 = *(const float*)(xb + (ja.y >> 16));
      float b0 = *(const float*)(xb + (jb.x & 0xffffu));
      float b1 = *(const float*)(xb + (jb.x >> 16));
      float b2 = *(const float*)(xb + (jb.y & 0xffffu));
      float b3 = *(const float*)(xb + (jb.y >> 16));
      float s0 = wa.x  * a0 + wa.y  * a1 + wa.z  * a2 + wa.w  * a3;
      float s1 = wb2.x * b0 + wb2.y * b1 + wb2.z * b2 + wb2.w * b3;
      As[r * LDA + col]        = (_Float16)s0;
      As[(r + 64) * LDA + col] = (_Float16)s1;
    }
  };

  // fragments: A row-per-lane (contiguous K), B col-per-lane (contiguous K)
  auto loadFrags = [&]() {
#pragma unroll
    for (int mt = 0; mt < MF; ++mt) {
      int row = wm + mt * 16 + lr;
      const v8h* p0 = (const v8h*)&As[row * LDA + hi * 16];
      const v8h* p1 = (const v8h*)&As[row * LDA + hi * 16 + 8];
      aF[mt] = cat8(*p0, *p1);
    }
#pragma unroll
    for (int nt = 0; nt < NF; ++nt) {
      int nrow = wn + nt * 16 + lr;
      const v8h* p0 = (const v8h*)&Bs[nrow * LDA + hi * 16];
      const v8h* p1 = (const v8h*)&Bs[nrow * LDA + hi * 16 + 8];
      bF[nt] = cat8(*p0, *p1);
    }
  };

  auto wmmaAll = [&]() {
#pragma unroll
    for (int mt = 0; mt < MF; ++mt)
#pragma unroll
      for (int nt = 0; nt < NF; ++nt)
        acc[mt][nt] = __builtin_amdgcn_wmma_f32_16x16x32_f16(
            false, aF[mt], false, bF[nt], (short)0, acc[mt][nt], false, false);
  };

  // ---- prologue: stage tile 0, grab its fragments ----
  asyncB(0);
  __syncthreads();          // metadata visible to all waves
  fillA(0);
  asm volatile("s_wait_asynccnt 0" ::: "memory");
  __syncthreads();          // tile 0 staged
  loadFrags();

  // ---- pipelined K loop: WMMA(t-1) co-executes with A-fill(t) ----
  for (int k0 = KS; k0 < KD; k0 += KS) {
    __syncthreads();        // every wave holds frags(t-1) in regs; LDS reusable
    asyncB(k0);             // DMA B(t) while computing
    wmmaAll();              // matrix pipe: tile t-1
    fillA(k0);              // VALU + gathers co-execute with WMMAs
    asm volatile("s_wait_asynccnt 0" ::: "memory");
    __syncthreads();        // tile t staged
    loadFrags();
  }
  wmmaAll();                // last tile

  // ---- store (NCHW; per-lane 8 consecutive hw -> b128-pair stores) ----
  float* __restrict__ outb = out + (size_t)bb * Nvalid * 4096;
#pragma unroll
  for (int mt = 0; mt < MF; ++mt) {
#pragma unroll
    for (int nt = 0; nt < NF; ++nt) {
      int n = n0 + wn + nt * 16 + lr;
      if (n < Nvalid) {
        int hwb = hw0 + wm + mt * 16 + hi * 8;
#pragma unroll
        for (int v = 0; v < 8; ++v) {
          float val = acc[mt][nt][v];
          if constexpr (!DEFORM) val += bias[n];
          outb[(size_t)n * 4096 + hwb + v] = val;
        }
      }
    }
  }
}

// ---------------------------------------------------------------------------
extern "C" void kernel_launch(void* const* d_in, const int* in_sizes, int n_in,
                              void* d_out, int out_size, void* d_ws, size_t ws_size,
                              hipStream_t stream) {
  (void)in_sizes; (void)n_in; (void)out_size; (void)ws_size;
  const float* x     = (const float*)d_in[0];   // (8,256,64,64)
  const float* w_off = (const float*)d_in[1];   // (18,256,3,3)
  const float* b_off = (const float*)d_in[2];   // (18,)
  const float* w_dcn = (const float*)d_in[3];   // (256,256,3,3)
  float*       out   = (float*)d_out;           // (8,256,64,64)

  // workspace layout
  char* ws = (char*)d_ws;
  _Float16* WtOff = (_Float16*)ws;                          // 32*2304 f16 = 147456 B
  _Float16* WtDcn = (_Float16*)(ws + 147456);               // 256*2304 f16 = 1179648 B
  float*    offBf = (float*)(ws + 147456 + 1179648);        // 8*18*4096 f32 = 2359296 B

  // 1) weight conversion to f16
  cvt_pad_woff<<<(32 * KD + 255) / 256, 256, 0, stream>>>(w_off, WtOff);
  cvt_wdcn<<<(256 * KD + 255) / 256, 256, 0, stream>>>(w_dcn, WtDcn, 256 * KD);

  // 2) offset conv (plain 3x3 conv as implicit-im2col WMMA GEMM, N padded 18->32)
  {
    size_t smem = B_LDS_BASE + (size_t)32 * LDA * 2;   // 40448 B
    dcn_wmma_gemm<32, false><<<dim3(MTOT / MT, 1), 256, smem, stream>>>(
        x, nullptr, WtOff, b_off, offBf, 18);
  }

  // 3) deformable sampling fused into main WMMA GEMM (M=32768, N=256, K=2304),
  //    single N tile -> bilinear im2col computed exactly once per output row
  {
    size_t smem = B_LDS_BASE + (size_t)256 * LDA * 2;  // 58368 B
    dcn_wmma_gemm<256, true><<<dim3(MTOT / MT, 1), 256, smem, stream>>>(
        x, offBf, WtDcn, nullptr, out, 256);
  }
}
